// EnhancedGovernanceAttention_4939212391178
// MI455X (gfx1250) — matrix operand order repaired and checked by reading, not compiled, not run on hardware
//
#include <hip/hip_runtime.h>
#include <stdint.h>

typedef __bf16 v16bf __attribute__((ext_vector_type(16)));
typedef float  v8f   __attribute__((ext_vector_type(8)));
typedef uint32_t v4u __attribute__((ext_vector_type(4)));
typedef uint32_t v8u __attribute__((ext_vector_type(8)));

#define NB  2
#define NS  2048
#define ND  2048
#define NH  16
#define NHD 128
#define NM  (NB * NS)         // 4096 rows of x
#define GSCALE 0.1f

union AFrag { v16bf v; uint32_t u[8]; uint16_t s[16]; };
union CFrag { v8f   v; float    f[8]; };

static __device__ __forceinline__ uint16_t f2bf(float x) {
  union { float f; uint32_t u; } c; c.f = x;
  uint32_t r = c.u + 0x7FFFu + ((c.u >> 16) & 1u);   // round-to-nearest-even
  return (uint16_t)(r >> 16);
}
static __device__ __forceinline__ float bf2f(uint16_t h) {
  union { uint32_t u; float f; } c; c.u = ((uint32_t)h) << 16;
  return c.f;
}

static __device__ __forceinline__ v8f wmma_bf16(AFrag a, AFrag b, v8f c) {
  return __builtin_amdgcn_wmma_f32_16x16x32_bf16(false, a.v, false, b.v,
                                                 (short)0, c, false, false);
}

// Low 32 bits of a flat shared-memory address are the LDS byte offset
// (see ISA 10.2 aperture calculation).
static __device__ __forceinline__ uint32_t lds_off(const void* p) {
  return (uint32_t)(uintptr_t)p;
}

// ---- CDNA5 async memory->LDS copy (ASYNCcnt path), 16 bytes per lane ----
static __device__ __forceinline__ void async_copy_b128(uint32_t dst_lds,
                                                       const void* src) {
  asm volatile("global_load_async_to_lds_b128 %0, %1, off"
               :: "v"(dst_lds), "v"(src) : "memory");
}
static __device__ __forceinline__ void wait_async0() {
  asm volatile("s_wait_asynccnt 0x0" ::: "memory");
}

// ---- CDNA5 Tensor Data Mover: 2D tile global->LDS (TENSORcnt path) ----
// D# per ISA ch.8: group0 {count=1, lds_addr, global_addr, type=2},
// group1 {data_size=2B, tensor/tile dims, dim0 stride}. tensor dims are set
// equal to the tile dims (global_addr is the tile origin) so OOB never fires.
static __device__ __forceinline__ void tdm_load_2d(uint32_t lds_addr,
                                                   const void* gaddr,
                                                   uint32_t tile_d0,
                                                   uint32_t tile_d1,
                                                   uint64_t stride_elems) {
  uint64_t ga = (uint64_t)(uintptr_t)gaddr;
  v4u g0;
  g0.x = 1u;                                       // count=1, user descriptor
  g0.y = lds_addr;                                 // LDS byte address
  g0.z = (uint32_t)ga;                             // global_addr[31:0]
  g0.w = ((uint32_t)(ga >> 32) & 0x01FFFFFFu) | (2u << 30);  // addr[56:32]|type=2
  v8u g1;
  g1.s0 = 0x00010000u;                             // data_size=1 (2 bytes)
  g1.s1 = (tile_d0 & 0xFFFFu) << 16;               // tensor_dim0[15:0] @48
  g1.s2 = (tile_d0 >> 16) | ((tile_d1 & 0xFFFFu) << 16);   // td0 hi | td1 lo
  g1.s3 = (tile_d1 >> 16) | (tile_d0 << 16);       // td1 hi | tile_dim0 @112
  g1.s4 = tile_d1 & 0xFFFFu;                       // tile_dim1 @128, tile_dim2=0
  g1.s5 = (uint32_t)stride_elems;                  // tensor_dim0_stride[31:0]
  g1.s6 = (uint32_t)(stride_elems >> 32) & 0xFFFFu;        // stride[47:32]
  g1.s7 = 0u;
  asm volatile("tensor_load_to_lds %0, %1" :: "s"(g0), "s"(g1) : "memory");
}
static __device__ __forceinline__ void wait_tensor0() {
  __builtin_amdgcn_s_wait_tensorcnt(0);
}

// reduction across the 16 lanes of a half-wave (rows of a C tile live there)
static __device__ __forceinline__ float hmax16(float x) {
  #pragma unroll
  for (int m = 1; m < 16; m <<= 1) x = fmaxf(x, __shfl_xor(x, m, 32));
  return x;
}
static __device__ __forceinline__ float hsum16(float x) {
  #pragma unroll
  for (int m = 1; m < 16; m <<= 1) x += __shfl_xor(x, m, 32);
  return x;
}

// ---------------- elementwise conversion kernels ----------------

__global__ void cvt_bf16_kernel(const float* __restrict__ in,
                                uint16_t* __restrict__ out, int n) {
  int i = blockIdx.x * 256 + threadIdx.x;
  if (i < n) out[i] = f2bf(in[i]);
}

// Wt[n*ND + k] = bf16(W[k*ND + n])   (so B-fragments read contiguous K)
__global__ void cvt_t_kernel(const float* __restrict__ W,
                             uint16_t* __restrict__ Wt) {
  int i = blockIdx.x * 256 + threadIdx.x;   // i = k*ND + n
  int n = i & (ND - 1);
  int k = i >> 11;
  Wt[(size_t)n * ND + k] = f2bf(W[i]);
}

// in-place RoPE on bf16 (B*S, D) tensor, head-dim pairs (i, i+64)
__global__ void rope_kernel(uint16_t* __restrict__ t) {
  int idx = blockIdx.x * 256 + threadIdx.x;   // NB*NS*NH*64 threads
  int i = idx & 63;
  int h = (idx >> 6) & 15;
  int s = (idx >> 10) & 2047;
  int b = idx >> 21;
  float inv = expf(-(float)(2 * i) * (9.210340371976184f / 128.0f));
  float ang = (float)s * inv;
  float cs = cosf(ang), sn = sinf(ang);
  uint16_t* p = t + ((size_t)(b * NS + s)) * ND + h * NHD;
  float x1 = bf2f(p[i]), x2 = bf2f(p[64 + i]);
  p[i]      = f2bf(x1 * cs - x2 * sn);
  p[64 + i] = f2bf(x2 * cs + x1 * sn);
}

// ---------------- bf16 WMMA GEMM ----------------
// C = A(MxK) * BT(NxK)^T.  Block = 4 waves = 64(M) x 64(N) tile.
// The shared 64x32 B panel is staged by the TDM (one tensor_load_to_lds
// per k-step, double-buffered so the DMA overlaps the WMMA work).

__global__ __launch_bounds__(128)
void gemm_bf16_kernel(const uint16_t* __restrict__ A,
                      const uint16_t* __restrict__ BT,
                      uint16_t* __restrict__ C, int M, int N, int K) {
  __shared__ __align__(16) uint16_t ldsB[2][64 * 32];

  const int lane = threadIdx.x & 31;
  const int wave = threadIdx.x >> 5;
  const int half = lane >> 4;
  const int l15  = lane & 15;
  const int tm  = blockIdx.y * 64 + wave * 16;
  const int tnb = blockIdx.x * 64;

  CFrag c[4];
  #pragma unroll
  for (int t = 0; t < 4; ++t)
    #pragma unroll
    for (int r = 0; r < 8; ++r) c[t].f[r] = 0.f;

  const uint16_t* arow = A + (size_t)(tm + l15) * K;
  const uint16_t* bpan = BT + (size_t)tnb * K;     // panel origin (row tnb)
  const int kba = half * 16;   // A-frag half-wave K base (0 / 16)
  const int kbb = half * 16;   // B-frag half-wave K base (0 / 16)
  const int nk = K / 32;

  if (wave == 0)
    tdm_load_2d(lds_off(&ldsB[0][0]), bpan, 32, 64, (uint64_t)K);

  for (int i = 0; i < nk; ++i) {
    if (wave == 0) wait_tensor0();
    __syncthreads();
    if (i + 1 < nk && wave == 0)
      tdm_load_2d(lds_off(&ldsB[(i + 1) & 1][0]), bpan + (i + 1) * 32,
                  32, 64, (uint64_t)K);

    const uint16_t* bbuf = ldsB[i & 1];
    const int k0 = i * 32;
    AFrag a;
    #pragma unroll
    for (int j = 0; j < 4; ++j) {
      a.u[j]     = *(const uint32_t*)(arow + k0 + kba + 2 * j);
      a.u[4 + j] = *(const uint32_t*)(arow + k0 + 8 + kba + 2 * j);
    }
    #pragma unroll
    for (int t = 0; t < 4; ++t) {
      AFrag bf;
      #pragma unroll
      for (int j = 0; j < 8; ++j)
        bf.u[j] = *(const uint32_t*)(bbuf + (t * 16 + l15) * 32 + kbb + 2 * j);
      c[t].v = wmma_bf16(a, bf, c[t].v);
    }
    __syncthreads();
  }

  #pragma unroll
  for (int t = 0; t < 4; ++t)
    #pragma unroll
    for (int r = 0; r < 8; ++r) {
      int m = tm + half * 8 + r;
      C[(size_t)m * N + tnb + t * 16 + l15] = f2bf(c[t].f[r]);
    }
}

// same GEMM but fp32 output + bias (final projection)
__global__ __launch_bounds__(128)
void gemm_out_kernel(const uint16_t* __restrict__ A,
                     const uint16_t* __restrict__ BT,
                     const float* __restrict__ bias,
                     float* __restrict__ C, int M, int N, int K) {
  __shared__ __align__(16) uint16_t ldsB[2][64 * 32];

  const int lane = threadIdx.x & 31;
  const int wave = threadIdx.x >> 5;
  const int half = lane >> 4;
  const int l15  = lane & 15;
  const int tm  = blockIdx.y * 64 + wave * 16;
  const int tnb = blockIdx.x * 64;

  CFrag c[4];
  #pragma unroll
  for (int t = 0; t < 4; ++t)
    #pragma unroll
    for (int r = 0; r < 8; ++r) c[t].f[r] = 0.f;

  const uint16_t* arow = A + (size_t)(tm + l15) * K;
  const uint16_t* bpan = BT + (size_t)tnb * K;
  const int kba = half * 16;
  const int kbb = half * 16;
  const int nk = K / 32;

  if (wave == 0)
    tdm_load_2d(lds_off(&ldsB[0][0]), bpan, 32, 64, (uint64_t)K);

  for (int i = 0; i < nk; ++i) {
    if (wave == 0) wait_tensor0();
    __syncthreads();
    if (i + 1 < nk && wave == 0)
      tdm_load_2d(lds_off(&ldsB[(i + 1) & 1][0]), bpan + (i + 1) * 32,
                  32, 64, (uint64_t)K);

    const uint16_t* bbuf = ldsB[i & 1];
    const int k0 = i * 32;
    AFrag a;
    #pragma unroll
    for (int j = 0; j < 4; ++j) {
      a.u[j]     = *(const uint32_t*)(arow + k0 + kba + 2 * j);
      a.u[4 + j] = *(const uint32_t*)(arow + k0 + 8 + kba + 2 * j);
    }
    #pragma unroll
    for (int t = 0; t < 4; ++t) {
      AFrag bf;
      #pragma unroll
      for (int j = 0; j < 8; ++j)
        bf.u[j] = *(const uint32_t*)(bbuf + (t * 16 + l15) * 32 + kbb + 2 * j);
      c[t].v = wmma_bf16(a, bf, c[t].v);
    }
    __syncthreads();
  }

  #pragma unroll
  for (int t = 0; t < 4; ++t) {
    float bn = bias[tnb + t * 16 + l15];
    #pragma unroll
    for (int r = 0; r < 8; ++r) {
      int m = tm + half * 8 + r;
      C[(size_t)m * N + tnb + t * 16 + l15] = c[t].f[r] + bn;
    }
  }
}

// ---------------- flash attention with governance bias ----------------
// block = 4 waves, each wave owns 16 q rows (64-row q block per (b,h)).
// K chunks staged by the TDM, V chunks by per-lane async copies;
// both double-buffered so the DMA engines overlap the WMMA/softmax work.

__global__ __launch_bounds__(128)
void attn_kernel(const uint16_t* __restrict__ q, const uint16_t* __restrict__ k,
                 const uint16_t* __restrict__ v, const float* __restrict__ pm,
                 const float* __restrict__ mem, uint16_t* __restrict__ out) {
  __shared__ __align__(16) uint16_t ldsK[2][32 * 128];
  __shared__ __align__(16) uint16_t ldsV[2][32 * 128];
  __shared__ __align__(16) uint16_t ldsP[4][16 * 32];

  const int lane = threadIdx.x & 31;
  const int wave = threadIdx.x >> 5;
  const int half = lane >> 4;
  const int l15  = lane & 15;
  const int b = blockIdx.z, h = blockIdx.y;
  const int qb = blockIdx.x * 64;
  const int qw = qb + wave * 16;
  const float scale = 0.08838834764831845f;   // 128^-0.5

  // preload this wave's 16x128 q tile as 4 A-fragments (K slices of 32)
  AFrag aq[4];
  {
    const uint16_t* qrow = q + ((size_t)(b * NS + qw + l15)) * ND + h * NHD;
    const int kba = half * 16;
    #pragma unroll
    for (int t = 0; t < 4; ++t) {
      #pragma unroll
      for (int j = 0; j < 4; ++j) {
        aq[t].u[j]     = *(const uint32_t*)(qrow + t * 32 + kba + 2 * j);
        aq[t].u[4 + j] = *(const uint32_t*)(qrow + t * 32 + 8 + kba + 2 * j);
      }
    }
  }

  float rm[8], rl[8];
  CFrag acc[8];   // 16x128 fp32 output accumulator
  #pragma unroll
  for (int r = 0; r < 8; ++r) { rm[r] = -__builtin_inff(); rl[r] = 0.f; }
  #pragma unroll
  for (int t = 0; t < 8; ++t)
    #pragma unroll
    for (int r = 0; r < 8; ++r) acc[t].f[r] = 0.f;

  const int nch = qb / 32 + 2;      // key chunks of 32 up to causal bound
  const int myqmax = qw + 15;
  const int kbb = half * 16;
  const int kba = half * 16;

  // per-thread async (V) staging addresses
  const int srow = threadIdx.x >> 2;
  const int sseg = (threadIdx.x & 3) * 32;
  const uint16_t* vsrc0 = v + ((size_t)(b * NS + srow)) * ND + h * NHD + sseg;
  const uint16_t* kchunk0 = k + ((size_t)(b * NS)) * ND + h * NHD;

  // stage chunk `ch` into buffer `buf`: TDM for K, async lane copies for V
  auto stage = [&](int ch, int buf) {
    const int ks = ch * 32;
    if (wave == 0)
      tdm_load_2d(lds_off(&ldsK[buf][0]), kchunk0 + (size_t)ks * ND,
                  128, 32, (uint64_t)ND);
    const size_t goff = (size_t)ks * ND;
    const uint32_t vdst = lds_off(&ldsV[buf][srow * 128 + sseg]);
    #pragma unroll
    for (int i = 0; i < 4; ++i)
      async_copy_b128(vdst + 16 * i, vsrc0 + goff + 8 * i);
  };

  stage(0, 0);

  for (int ch = 0; ch < nch; ++ch) {
    if (wave == 0) wait_tensor0();
    wait_async0();
    __syncthreads();
    if (ch + 1 < nch) stage(ch + 1, (ch + 1) & 1);

    const int ks = ch * 32;
    const uint16_t* Kb = ldsK[ch & 1];
    const uint16_t* Vb = ldsV[ch & 1];

    if (ks <= myqmax) {
      // prefetch next chunk's policy-mask rows (dominant HBM stream)
      if (ks + 32 < NS) {
        int qp = qw + half * 8;
        __builtin_prefetch(pm + ((size_t)(h * NS + qp)) * NS + ks + 32 + l15, 0, 1);
      }

      // scores: q(16x128) @ k_chunk(32x128)^T -> two 16x16 tiles
      CFrag s0, s1;
      #pragma unroll
      for (int r = 0; r < 8; ++r) { s0.f[r] = 0.f; s1.f[r] = 0.f; }
      #pragma unroll
      for (int t = 0; t < 4; ++t) {
        AFrag b0, b1;
        #pragma unroll
        for (int j = 0; j < 8; ++j) {
          b0.u[j] = *(const uint32_t*)(Kb + (l15)*128      + t * 32 + kbb + 2 * j);
          b1.u[j] = *(const uint32_t*)(Kb + (16 + l15)*128 + t * 32 + kbb + 2 * j);
        }
        s0.v = wmma_bf16(aq[t], b0, s0.v);
        s1.v = wmma_bf16(aq[t], b1, s1.v);
      }

      // bias + causal mask + online softmax (row lives in 16 lanes of half)
      #pragma unroll
      for (int r = 0; r < 8; ++r) {
        int qpos = qw + half * 8 + r;
        int kp0 = ks + l15;
        int kp1 = ks + 16 + l15;
        float b0 = pm[((size_t)(h * NS + qpos)) * NS + kp0] * GSCALE
                   + log1pf(mem[b * NS + kp0] * GSCALE + 1e-8f);
        float b1 = pm[((size_t)(h * NS + qpos)) * NS + kp1] * GSCALE
                   + log1pf(mem[b * NS + kp1] * GSCALE + 1e-8f);
        float v0 = s0.f[r] * scale + b0;
        float v1 = s1.f[r] * scale + b1;
        if (kp0 > qpos) v0 = -__builtin_inff();
        if (kp1 > qpos) v1 = -__builtin_inff();
        float mnew = fmaxf(rm[r], hmax16(fmaxf(v0, v1)));
        float p0 = expf(v0 - mnew);
        float p1 = expf(v1 - mnew);
        float tsum = hsum16(p0 + p1);
        float corr = expf(rm[r] - mnew);
        rl[r] = rl[r] * corr + tsum;
        rm[r] = mnew;
        #pragma unroll
        for (int t = 0; t < 8; ++t) acc[t].f[r] *= corr;
        int rowl = half * 8 + r;                 // C-layout -> LDS (row-major P)
        ldsP[wave][rowl * 32 + l15]      = f2bf(p0);
        ldsP[wave][rowl * 32 + 16 + l15] = f2bf(p1);
      }

      // P(16x32) as A-fragment via LDS round trip (same-wave, DS in-order)
      AFrag ap;
      #pragma unroll
      for (int j = 0; j < 4; ++j) {
        ap.u[j]     = *(const uint32_t*)(&ldsP[wave][l15 * 32 + kba + 2 * j]);
        ap.u[4 + j] = *(const uint32_t*)(&ldsP[wave][l15 * 32 + 8 + kba + 2 * j]);
      }
      // acc += P @ V_chunk(32x128)
      #pragma unroll
      for (int t = 0; t < 8; ++t) {
        AFrag bv;
        #pragma unroll
        for (int j = 0; j < 8; ++j) {
          int key = kbb + 2 * j;
          bv.s[2 * j]     = Vb[(size_t)key * 128       + t * 16 + l15];
          bv.s[2 * j + 1] = Vb[(size_t)(key + 1) * 128 + t * 16 + l15];
        }
        acc[t].v = wmma_bf16(ap, bv, acc[t].v);
      }
    }
    __syncthreads();
  }

  // normalize + store (B,S,H,HD) flattened as (B*S, D) bf16
  #pragma unroll
  for (int r = 0; r < 8; ++r) {
    int qpos = qw + half * 8 + r;
    float inv = 1.0f / rl[r];
    #pragma unroll
    for (int t = 0; t < 8; ++t)
      out[((size_t)(b * NS + qpos)) * ND + h * NHD + t * 16 + l15] =
          f2bf(acc[t].f[r] * inv);
  }
}

// ---------------- host-side launch ----------------

extern "C" void kernel_launch(void* const* d_in, const int* in_sizes, int n_in,
                              void* d_out, int out_size, void* d_ws, size_t ws_size,
                              hipStream_t stream) {
  (void)in_sizes; (void)n_in; (void)out_size; (void)ws_size;
  const float* x   = (const float*)d_in[0];
  const float* Wq  = (const float*)d_in[1];
  const float* Wk  = (const float*)d_in[2];
  const float* Wv  = (const float*)d_in[3];
  const float* Wo  = (const float*)d_in[4];
  const float* bo  = (const float*)d_in[5];
  const float* pm  = (const float*)d_in[6];
  const float* mem = (const float*)d_in[7];
  float* out = (float*)d_out;

  char* ws = (char*)d_ws;
  uint16_t* xbf = (uint16_t*)(ws);                    // 16 MB
  uint16_t* wqT = (uint16_t*)(ws + (16ull << 20));    //  8 MB each
  uint16_t* wkT = (uint16_t*)(ws + (24ull << 20));
  uint16_t* wvT = (uint16_t*)(ws + (32ull << 20));
  uint16_t* woT = (uint16_t*)(ws + (40ull << 20));
  uint16_t* qb  = (uint16_t*)(ws + (48ull << 20));    // 16 MB each
  uint16_t* kb  = (uint16_t*)(ws + (64ull << 20));
  uint16_t* vb  = (uint16_t*)(ws + (80ull << 20));
  uint16_t* ob  = (uint16_t*)(ws + (96ull << 20));

  const int nx = NM * ND;
  cvt_bf16_kernel<<<nx / 256, 256, 0, stream>>>(x, xbf, nx);
  cvt_t_kernel<<<(ND * ND) / 256, 256, 0, stream>>>(Wq, wqT);
  cvt_t_kernel<<<(ND * ND) / 256, 256, 0, stream>>>(Wk, wkT);
  cvt_t_kernel<<<(ND * ND) / 256, 256, 0, stream>>>(Wv, wvT);
  cvt_t_kernel<<<(ND * ND) / 256, 256, 0, stream>>>(Wo, woT);

  dim3 gg(ND / 64, NM / 64);
  gemm_bf16_kernel<<<gg, 128, 0, stream>>>(xbf, wqT, qb, NM, ND, ND);
  gemm_bf16_kernel<<<gg, 128, 0, stream>>>(xbf, wkT, kb, NM, ND, ND);
  gemm_bf16_kernel<<<gg, 128, 0, stream>>>(xbf, wvT, vb, NM, ND, ND);

  const int nr = NB * NS * NH * 64;
  rope_kernel<<<nr / 256, 256, 0, stream>>>(qb);
  rope_kernel<<<nr / 256, 256, 0, stream>>>(kb);

  dim3 ga(NS / 64, NH, NB);
  attn_kernel<<<ga, 128, 0, stream>>>(qb, kb, vb, pm, mem, ob);

  gemm_out_kernel<<<gg, 128, 0, stream>>>(ob, woT, bo, out, NM, ND, ND);
}
